// STGNN_59846074303162
// MI455X (gfx1250) — matrix operand order repaired and checked by reading, not compiled, not run on hardware
//
#include <hip/hip_runtime.h>
#include <math.h>

// ---------------------------------------------------------------------------
// STGNN on gfx1250: all GEMMs via v_wmma_f32_16x16x32_f16 (f32 accumulate).
// Weights are pre-converted to f16 (row-major [Nc,K]) once per launch so
// B-fragments are raw 32B v16h loads (no per-tile v_cvt traffic).
// T=4, N=50000, E=400000, IN=64, H=128, ED=10
// ---------------------------------------------------------------------------

#define T_STEPS 4
#define NN      50000
#define EE      400000
#define IN_DIM  64
#define HH      128
#define ED_DIM  10

typedef __attribute__((ext_vector_type(16))) _Float16 v16h;
typedef __attribute__((ext_vector_type(8)))  float    v8f;

// --- WMMA helper ------------------------------------------------------------
__device__ __forceinline__ v8f wmma_f16(v16h a, v16h b, v8f c) {
  return __builtin_amdgcn_wmma_f32_16x16x32_f16(
      /*neg_a=*/false, a, /*neg_b=*/false, b,
      /*c_mod=*/(short)0, c, /*reuse_a=*/false, /*reuse_b=*/false);
}

// A-fragment (16x32 f16, MxK): lane L (m = L&15, hs = L>>4) holds
//   e in [0,15] -> K = k0 + e + 16*hs   (contiguous 16 floats)
__device__ __forceinline__ v16h load_afrag(const float* __restrict__ arow,
                                           int k0, int hs, float scale) {
  const float4* p0 = (const float4*)(arow + k0 + 16 * hs);
  const float4* p1 = (const float4*)(arow + k0 + 16 * hs + 8);
  float4 a0 = p0[0], a1 = p0[1];
  float4 a2 = p1[0], a3 = p1[1];
  v16h r;
  r[0]  = (_Float16)(a0.x * scale); r[1]  = (_Float16)(a0.y * scale);
  r[2]  = (_Float16)(a0.z * scale); r[3]  = (_Float16)(a0.w * scale);
  r[4]  = (_Float16)(a1.x * scale); r[5]  = (_Float16)(a1.y * scale);
  r[6]  = (_Float16)(a1.z * scale); r[7]  = (_Float16)(a1.w * scale);
  r[8]  = (_Float16)(a2.x * scale); r[9]  = (_Float16)(a2.y * scale);
  r[10] = (_Float16)(a2.z * scale); r[11] = (_Float16)(a2.w * scale);
  r[12] = (_Float16)(a3.x * scale); r[13] = (_Float16)(a3.y * scale);
  r[14] = (_Float16)(a3.z * scale); r[15] = (_Float16)(a3.w * scale);
  return r;
}

// B-fragment (32x16 f16, KxN) for B = W^T with W16[Nc,K] row-major f16:
// lane L holds column n = L&15, K = k0 + e + 16*hs -> one 32B v16h load.
__device__ __forceinline__ v16h load_bfrag16(const _Float16* __restrict__ W16,
                                             int nrow, int K, int k0, int hs) {
  return *(const v16h*)(W16 + (size_t)nrow * K + k0 + 16 * hs);
}

__device__ __forceinline__ float sigf(float x) { return 1.f / (1.f + __expf(-x)); }

// --- f32 -> f16 weight pre-conversion ---------------------------------------
__global__ void k_cvt16(const float* __restrict__ s, _Float16* __restrict__ d, int n) {
  int i = blockIdx.x * blockDim.x + threadIdx.x;
  if (i < n) d[i] = (_Float16)s[i];
}

// --- generic GEMM: C[M,Nc] = act(A[M,K] @ W[Nc,K]^T + b), act: 0=none 1=relu
__global__ void k_gemm(const float* __restrict__ A, const _Float16* __restrict__ W16,
                       const float* __restrict__ bias, float* __restrict__ C,
                       int M, int K, int Nc, int act) {
  int lane = threadIdx.x & 31, wid = threadIdx.x >> 5;
  int mtile = blockIdx.x * (blockDim.x >> 5) + wid;
  if (mtile * 16 >= M) return;                  // whole-wave exit, EXEC stays full
  int hs = lane >> 4, l15 = lane & 15;
  int nb = blockIdx.y * 64;
  const float* arow = A + (size_t)(mtile * 16 + l15) * K;
  v8f c0 = {}, c1 = {}, c2 = {}, c3 = {};
  for (int k0 = 0; k0 < K; k0 += 32) {
    v16h a = load_afrag(arow, k0, hs, 1.f);
    c0 = wmma_f16(a, load_bfrag16(W16, nb + l15,      K, k0, hs), c0);
    c1 = wmma_f16(a, load_bfrag16(W16, nb + 16 + l15, K, k0, hs), c1);
    c2 = wmma_f16(a, load_bfrag16(W16, nb + 32 + l15, K, k0, hs), c2);
    c3 = wmma_f16(a, load_bfrag16(W16, nb + 48 + l15, K, k0, hs), c3);
  }
  v8f cc[4] = {c0, c1, c2, c3};
#pragma unroll
  for (int s = 0; s < 4; s++) {
    int n = nb + s * 16 + l15;
    float bv = bias[n];
#pragma unroll
    for (int i = 0; i < 8; i++) {
      int m = mtile * 16 + i + 8 * hs;
      float v = cc[s][i] + bv;
      if (act) v = fmaxf(v, 0.f);
      C[(size_t)m * Nc + n] = v;
    }
  }
}

// --- fused message GEMM + edge-weight scale + atomic mean-scatter -----------
// msg = relu(X[src] @ mW^T + mb) * ew ; agg[dst] += msg   (H=K=128)
__global__ void k_msg(const float* __restrict__ X, const int* __restrict__ src,
                      const int* __restrict__ dst, const float* __restrict__ ew,
                      const _Float16* __restrict__ mW16, const float* __restrict__ mb,
                      float* __restrict__ agg, int E) {
  const int K = HH;
  int lane = threadIdx.x & 31, wid = threadIdx.x >> 5;
  int mtile = blockIdx.x * (blockDim.x >> 5) + wid;
  if (mtile * 16 >= E) return;
  int hs = lane >> 4, l15 = lane & 15;
  int nb = blockIdx.y * 64;
  int srow = src[mtile * 16 + l15];             // gather (x is L2-resident)
  const float* arow = X + (size_t)srow * K;
  v8f c0 = {}, c1 = {}, c2 = {}, c3 = {};
  for (int k0 = 0; k0 < K; k0 += 32) {
    v16h a = load_afrag(arow, k0, hs, 1.f);
    c0 = wmma_f16(a, load_bfrag16(mW16, nb + l15,      K, k0, hs), c0);
    c1 = wmma_f16(a, load_bfrag16(mW16, nb + 16 + l15, K, k0, hs), c1);
    c2 = wmma_f16(a, load_bfrag16(mW16, nb + 32 + l15, K, k0, hs), c2);
    c3 = wmma_f16(a, load_bfrag16(mW16, nb + 48 + l15, K, k0, hs), c3);
  }
  int d8[8]; float w8[8];
#pragma unroll
  for (int i = 0; i < 8; i++) {
    int m = mtile * 16 + i + 8 * hs;
    d8[i] = dst[m]; w8[i] = ew[m];
  }
  v8f cc[4] = {c0, c1, c2, c3};
#pragma unroll
  for (int s = 0; s < 4; s++) {
    int n = nb + s * 16 + l15;
    float bv = mb[n];
#pragma unroll
    for (int i = 0; i < 8; i++) {
      float v = fmaxf(cc[s][i] + bv, 0.f) * w8[i];
      atomicAdd(&agg[(size_t)d8[i] * HH + n], v);
    }
  }
}

// --- fused GRU cell (K = H = 128). rdeg != nullptr folds mean into X load.
// gi = (X*rdeg) @ Wih^T + bih ; gh = Hin @ Whh^T + bhh
// r=sig(gi_r+gh_r), z=sig(gi_z+gh_z), n=tanh(gi_n + r*gh_n)
// out = (1-z)*n + z*Hin
__global__ void k_gru(const float* __restrict__ X, const float* __restrict__ Hin,
                      const float* __restrict__ rdeg,
                      const _Float16* __restrict__ Wih16,
                      const _Float16* __restrict__ Whh16,
                      const float* __restrict__ bih, const float* __restrict__ bhh,
                      float* __restrict__ Out, int M) {
  const int K = HH;
  int lane = threadIdx.x & 31, wid = threadIdx.x >> 5;
  int mtile = blockIdx.x * (blockDim.x >> 5) + wid;
  if (mtile * 16 >= M) return;
  int hs = lane >> 4, l15 = lane & 15;
  int n = blockIdx.y * 16 + l15;                // output column in [0,H)
  int arowi = mtile * 16 + l15;
  float sc = rdeg ? rdeg[arowi] : 1.f;
  const float* xrow = X   + (size_t)arowi * K;
  const float* hrow = Hin + (size_t)arowi * K;
  v8f gir = {}, giz = {}, gin = {}, ghr = {}, ghz = {}, ghn = {};
  for (int k0 = 0; k0 < K; k0 += 32) {
    v16h ax = load_afrag(xrow, k0, hs, sc);
    v16h ah = load_afrag(hrow, k0, hs, 1.f);
    gir = wmma_f16(ax, load_bfrag16(Wih16, 0 * HH + n, K, k0, hs), gir);
    giz = wmma_f16(ax, load_bfrag16(Wih16, 1 * HH + n, K, k0, hs), giz);
    gin = wmma_f16(ax, load_bfrag16(Wih16, 2 * HH + n, K, k0, hs), gin);
    ghr = wmma_f16(ah, load_bfrag16(Whh16, 0 * HH + n, K, k0, hs), ghr);
    ghz = wmma_f16(ah, load_bfrag16(Whh16, 1 * HH + n, K, k0, hs), ghz);
    ghn = wmma_f16(ah, load_bfrag16(Whh16, 2 * HH + n, K, k0, hs), ghn);
  }
  float br = bih[n], bz = bih[HH + n], bn = bih[2 * HH + n];
  float cr = bhh[n], cz = bhh[HH + n], cn = bhh[2 * HH + n];
#pragma unroll
  for (int i = 0; i < 8; i++) {
    int m = mtile * 16 + i + 8 * hs;
    float r  = sigf(gir[i] + br + ghr[i] + cr);
    float z  = sigf(giz[i] + bz + ghz[i] + cz);
    float nn = tanhf(gin[i] + bn + r * (ghn[i] + cn));
    float h  = Hin[(size_t)m * HH + n];
    Out[(size_t)m * HH + n] = (1.f - z) * nn + z * h;
  }
}

// --- edge-weight MLP: ew = sigmoid(relu(ea@ep1^T+b1)@ep2^T+b2)  (K=10) ------
__global__ void k_edgew(const float* __restrict__ ea, const float* __restrict__ w1,
                        const float* __restrict__ b1, const float* __restrict__ w2,
                        const float* __restrict__ b2, float* __restrict__ ew, int E) {
  __shared__ float sw1[64 * ED_DIM];
  __shared__ float sb1[64];
  __shared__ float sw2[64];
  __shared__ float sb2;
  for (int i = threadIdx.x; i < 64 * ED_DIM; i += blockDim.x) sw1[i] = w1[i];
  for (int i = threadIdx.x; i < 64; i += blockDim.x) { sb1[i] = b1[i]; sw2[i] = w2[i]; }
  if (threadIdx.x == 0) sb2 = b2[0];
  __syncthreads();
  int e = blockIdx.x * blockDim.x + threadIdx.x;
  if (e >= E) return;
  float a[ED_DIM];
  const float* p = ea + (size_t)e * ED_DIM;
#pragma unroll
  for (int k = 0; k < ED_DIM; k++) a[k] = p[k];
  float acc = sb2;
#pragma unroll 4
  for (int j = 0; j < 64; j++) {
    float t = sb1[j];
#pragma unroll
    for (int k = 0; k < ED_DIM; k++) t += sw1[j * ED_DIM + k] * a[k];
    acc += fmaxf(t, 0.f) * sw2[j];
  }
  ew[e] = sigf(acc);
}

// --- misc small kernels ------------------------------------------------------
__global__ void k_zero(float* __restrict__ p, int n) {
  int i = blockIdx.x * blockDim.x + threadIdx.x;
  if (i < n) p[i] = 0.f;
}
__global__ void k_deg(const int* __restrict__ dst, float* __restrict__ deg, int E) {
  int e = blockIdx.x * blockDim.x + threadIdx.x;
  if (e < E) atomicAdd(&deg[dst[e]], 1.f);
}
__global__ void k_rdeg(float* __restrict__ deg, int n) {  // in-place -> 1/clip(deg,1)
  int i = blockIdx.x * blockDim.x + threadIdx.x;
  if (i < n) deg[i] = 1.f / fmaxf(deg[i], 1.f);
}
__global__ void k_rownorm(float* __restrict__ h, int M) { // L2-normalize rows (H=128)
  int lane = threadIdx.x & 31, wid = threadIdx.x >> 5;
  int row = blockIdx.x * (blockDim.x >> 5) + wid;
  if (row >= M) return;
  float* p = h + (size_t)row * HH;
  float v0 = p[lane], v1 = p[lane + 32], v2 = p[lane + 64], v3 = p[lane + 96];
  float s = v0 * v0 + v1 * v1 + v2 * v2 + v3 * v3;
#pragma unroll
  for (int m = 16; m >= 1; m >>= 1) s += __shfl_xor(s, m, 32);
  float inv = 1.f / fmaxf(sqrtf(s), 1e-12f);
  p[lane] = v0 * inv; p[lane + 32] = v1 * inv;
  p[lane + 64] = v2 * inv; p[lane + 96] = v3 * inv;
}
// final head: out[N,3] = tmp[N,128] @ regW[3,128]^T + regb
__global__ void k_reg(const float* __restrict__ tmp, const float* __restrict__ rw,
                      const float* __restrict__ rb, float* __restrict__ out, int M) {
  int row = blockIdx.x * blockDim.x + threadIdx.x;
  if (row >= M) return;
  const float* t = tmp + (size_t)row * HH;
  float s0 = rb[0], s1 = rb[1], s2 = rb[2];
  for (int k = 0; k < HH; k++) {
    float x = t[k];
    s0 += x * rw[k]; s1 += x * rw[HH + k]; s2 += x * rw[2 * HH + k];
  }
  out[(size_t)row * 3 + 0] = s0;
  out[(size_t)row * 3 + 1] = s1;
  out[(size_t)row * 3 + 2] = s2;
}

// ---------------------------------------------------------------------------
extern "C" void kernel_launch(void* const* d_in, const int* in_sizes, int n_in,
                              void* d_out, int out_size, void* d_ws, size_t ws_size,
                              hipStream_t stream) {
  const float* xs    = (const float*)d_in[0];
  const int*   ei    = (const int*)  d_in[1];
  const float* ea    = (const float*)d_in[2];
  const float* pre_W = (const float*)d_in[3];  const float* pre_b = (const float*)d_in[4];
  const float* ep1_W = (const float*)d_in[5];  const float* ep1_b = (const float*)d_in[6];
  const float* ep2_W = (const float*)d_in[7];  const float* ep2_b = (const float*)d_in[8];
  const float* l1mW  = (const float*)d_in[9];  const float* l1mb  = (const float*)d_in[10];
  const float* l1Wih = (const float*)d_in[11]; const float* l1Whh = (const float*)d_in[12];
  const float* l1bih = (const float*)d_in[13]; const float* l1bhh = (const float*)d_in[14];
  const float* l2mW  = (const float*)d_in[15]; const float* l2mb  = (const float*)d_in[16];
  const float* l2Wih = (const float*)d_in[17]; const float* l2Whh = (const float*)d_in[18];
  const float* l2bih = (const float*)d_in[19]; const float* l2bhh = (const float*)d_in[20];
  const float* tgWih = (const float*)d_in[21]; const float* tgWhh = (const float*)d_in[22];
  const float* tgbih = (const float*)d_in[23]; const float* tgbhh = (const float*)d_in[24];
  const float* postW = (const float*)d_in[25]; const float* postb = (const float*)d_in[26];
  const float* regW  = (const float*)d_in[27]; const float* regb  = (const float*)d_in[28];
  float* out = (float*)d_out;

  // ---- workspace carve-up: f32 buffers (~233 MB) then f16 weights (~0.7 MB)
  float* W = (float*)d_ws;
  size_t o = 0;
  float* ew   = W + o; o += (size_t)EE;
  float* deg  = W + o; o += (size_t)NN;
  float* agg  = W + o; o += (size_t)NN * HH;   // also reused as post-MLP tmp
  float* bufA = W + o; o += (size_t)NN * HH;
  float* bufB = W + o; o += (size_t)NN * HH;
  float* hA   = W + o; o += (size_t)NN * HH;
  float* hB   = W + o; o += (size_t)NN * HH;
  float* hseq = W + o; o += (size_t)T_STEPS * NN * HH;
  o = (o + 15) & ~(size_t)15;                  // 64B-align the f16 region
  _Float16* h16 = (_Float16*)(W + o);
  size_t ho = 0;
  _Float16* preW16  = h16 + ho; ho += (size_t)HH * IN_DIM;       // 128x64
  _Float16* l1mW16  = h16 + ho; ho += (size_t)HH * HH;           // 128x128
  _Float16* l2mW16  = h16 + ho; ho += (size_t)HH * HH;
  _Float16* l1Wih16 = h16 + ho; ho += (size_t)3 * HH * HH;       // 384x128
  _Float16* l1Whh16 = h16 + ho; ho += (size_t)3 * HH * HH;
  _Float16* l2Wih16 = h16 + ho; ho += (size_t)3 * HH * HH;
  _Float16* l2Whh16 = h16 + ho; ho += (size_t)3 * HH * HH;
  _Float16* tgWih16 = h16 + ho; ho += (size_t)3 * HH * HH;
  _Float16* tgWhh16 = h16 + ho; ho += (size_t)3 * HH * HH;
  _Float16* postW16 = h16 + ho; ho += (size_t)HH * HH;
  (void)ws_size; (void)in_sizes; (void)n_in; (void)out_size;

  const int TPB = 256, WPB = 8;                       // 8 wave32 per block
  const int mtN = (NN + 15) / 16, mtE = (EE + 15) / 16;
  dim3 gPre((mtN + WPB - 1) / WPB, HH / 64);          // 16x64 tiles
  dim3 gMsg((mtE + WPB - 1) / WPB, HH / 64);
  dim3 gGru((mtN + WPB - 1) / WPB, HH / 16);          // 16x16 tiles, 6 accums
  int  gZa  = (NN * HH + TPB - 1) / TPB;
  int  gE   = (EE + TPB - 1) / TPB;
  int  gN   = (NN + TPB - 1) / TPB;
  int  gRn  = (NN + WPB - 1) / WPB;

  // ---- pre-convert weights to f16 (tiny; once per launch, deterministic)
  {
    struct { const float* s; _Float16* d; int n; } cv[] = {
      {pre_W, preW16, HH * IN_DIM},
      {l1mW,  l1mW16,  HH * HH}, {l2mW, l2mW16, HH * HH},
      {l1Wih, l1Wih16, 3 * HH * HH}, {l1Whh, l1Whh16, 3 * HH * HH},
      {l2Wih, l2Wih16, 3 * HH * HH}, {l2Whh, l2Whh16, 3 * HH * HH},
      {tgWih, tgWih16, 3 * HH * HH}, {tgWhh, tgWhh16, 3 * HH * HH},
      {postW, postW16, HH * HH},
    };
    for (int i = 0; i < 10; i++)
      k_cvt16<<<(cv[i].n + TPB - 1) / TPB, TPB, 0, stream>>>(cv[i].s, cv[i].d, cv[i].n);
  }

  for (int t = 0; t < T_STEPS; t++) {
    const float* x_t  = xs + (size_t)t * NN * IN_DIM;
    const int*   src  = ei + (size_t)t * 2 * EE;
    const int*   dst  = src + EE;
    const float* ea_t = ea + (size_t)t * EE * ED_DIM;
    float* hs_t = hseq + (size_t)t * NN * HH;

    // xt = relu(x_t @ pre_W^T + pre_b)
    k_gemm<<<gPre, TPB, 0, stream>>>(x_t, preW16, pre_b, bufA, NN, IN_DIM, HH, 1);
    // edge weights + degrees
    k_edgew<<<gE, TPB, 0, stream>>>(ea_t, ep1_W, ep1_b, ep2_W, ep2_b, ew, EE);
    k_zero<<<gN, TPB, 0, stream>>>(deg, NN);
    k_deg<<<gE, TPB, 0, stream>>>(dst, deg, EE);
    k_rdeg<<<gN, TPB, 0, stream>>>(deg, NN);
    // layer 1: bufA -> bufB
    k_zero<<<gZa, TPB, 0, stream>>>(agg, NN * HH);
    k_msg<<<gMsg, TPB, 0, stream>>>(bufA, src, dst, ew, l1mW16, l1mb, agg, EE);
    k_gru<<<gGru, TPB, 0, stream>>>(agg, bufA, deg, l1Wih16, l1Whh16, l1bih, l1bhh, bufB, NN);
    k_rownorm<<<gRn, TPB, 0, stream>>>(bufB, NN);
    // layer 2: bufB -> hseq[t]
    k_zero<<<gZa, TPB, 0, stream>>>(agg, NN * HH);
    k_msg<<<gMsg, TPB, 0, stream>>>(bufB, src, dst, ew, l2mW16, l2mb, agg, EE);
    k_gru<<<gGru, TPB, 0, stream>>>(agg, bufB, deg, l2Wih16, l2Whh16, l2bih, l2bhh, hs_t, NN);
    k_rownorm<<<gRn, TPB, 0, stream>>>(hs_t, NN);
  }

  // temporal GRU over hseq, h0 = 0
  k_zero<<<gZa, TPB, 0, stream>>>(hA, NN * HH);
  float* cur = hA; float* nxt = hB;
  for (int t = 0; t < T_STEPS; t++) {
    const float* hs_t = hseq + (size_t)t * NN * HH;
    k_gru<<<gGru, TPB, 0, stream>>>(hs_t, cur, nullptr, tgWih16, tgWhh16, tgbih, tgbhh, nxt, NN);
    float* tmp = cur; cur = nxt; nxt = tmp;
  }
  // head: relu(hT @ postW^T + postb) @ regW^T + regb
  k_gemm<<<gPre, TPB, 0, stream>>>(cur, postW16, postb, agg, NN, HH, HH, 1);
  k_reg<<<gN, TPB, 0, stream>>>(agg, regW, regb, out, NN);
}
